// NeuralCircuitLTC_49658411876619
// MI455X (gfx1250) — matrix operand order repaired and checked by reading, not compiled, not run on hardware
//
#include <hip/hip_runtime.h>
#include <hip/hip_bf16.h>
#include <math.h>

// Problem dims (fixed by reference)
#define BB   64
#define TT   256
#define NIN  1024
#define HH   2048
#define MM   512
#define LLAY 2
#define HM   (HH + MM)

typedef __attribute__((ext_vector_type(16))) __bf16 v16bf;
typedef __attribute__((ext_vector_type(8)))  __bf16 v8bf;
typedef __attribute__((ext_vector_type(8)))  float  v8f;

// ---------------------------------------------------------------------------
// Elementwise helpers
// ---------------------------------------------------------------------------
__global__ void k_f32_to_bf16(const float* __restrict__ in, __bf16* __restrict__ out, int n) {
  int i = blockIdx.x * blockDim.x + threadIdx.x;
  if (i < n) out[i] = (__bf16)in[i];
}

__global__ void k_add_to_bf16(const float* __restrict__ a, const float* __restrict__ b,
                              __bf16* __restrict__ out, int n) {
  int i = blockIdx.x * blockDim.x + threadIdx.x;
  if (i < n) out[i] = (__bf16)(a[i] + b[i]);
}

__global__ void k_add_f32(const float* __restrict__ a, const float* __restrict__ b,
                          float* __restrict__ out, int n) {
  int i = blockIdx.x * blockDim.x + threadIdx.x;
  if (i < n) out[i] = a[i] + b[i];
}

// ---------------------------------------------------------------------------
// Generic bf16 WMMA GEMM, 2x2 tile-blocked (32x32 output per wave):
//   out[r,c] = scale*( A1[r,:K1]@W[c,0:K1]^T + A2[r,:K2]@W[c,K1:K1+K2]^T + bias[c] )
//              (+ addsrc[r,c])
// A1 row stride = K1, A2 row stride = K2, W row stride = ldW (W row index = out col).
// Per k-step: 2 A fragments + 2 B fragments (8 x b128 loads), 4 independent
// V_WMMA_F32_16X16X32_BF16 accumulation chains. Fragment loads follow the
// ISA 7.12.2 VGPR layouts (two 16B contiguous runs per operand per lane).
// rowT2/colT2 are tile counts in units of 32.
// ---------------------------------------------------------------------------
static __device__ inline v16bf ld16(const __bf16* p0, const __bf16* p1) {
  v8bf lo = *(const v8bf*)p0;
  v8bf hi = *(const v8bf*)p1;
  v16bf r;
#pragma unroll
  for (int i = 0; i < 8; ++i) { r[i] = lo[i]; r[i + 8] = hi[i]; }
  return r;
}

__global__ void k_gemm(const __bf16* __restrict__ A1, int K1,
                       const __bf16* __restrict__ A2, int K2,
                       const __bf16* __restrict__ W, int ldW,
                       const float* __restrict__ bias, float scale,
                       const float* __restrict__ addsrc, int addld,
                       float* __restrict__ out, int outld,
                       int rowT2, int colT2) {
  int wave = blockIdx.x * (blockDim.x >> 5) + (threadIdx.x >> 5);
  int total = rowT2 * colT2;
  if (wave >= total) return;
  int lane = threadIdx.x & 31;
  int r    = lane & 15;         // row-in-tile for A, col-in-tile for B/C
  int half = lane >> 4;
  int tm = wave / colT2;
  int tn = wave - tm * colT2;
  int rowBase = tm << 5, colBase = tn << 5;

  v8f acc00 = {}, acc01 = {}, acc10 = {}, acc11 = {};

  // Segment 1: K1 over A1 and W[:, 0:K1]
  {
    const __bf16* arow0 = A1 + (size_t)(rowBase + r) * K1;
    const __bf16* arow1 = arow0 + (size_t)16 * K1;
    const __bf16* wrow0 = W + (size_t)(colBase + r) * ldW;
    const __bf16* wrow1 = wrow0 + (size_t)16 * ldW;
    for (int kb = 0; kb < K1; kb += 32) {
      int ao = kb + half * 8;   // A: K in {h*8..+8} and {16+h*8..+8}
      int bo = kb + half * 8;   // B: K in {h*8..+8} and {16+h*8..+8}
      v16bf a0 = ld16(arow0 + ao, arow0 + ao + 16);
      v16bf a1 = ld16(arow1 + ao, arow1 + ao + 16);
      v16bf b0 = ld16(wrow0 + bo, wrow0 + bo + 16);
      v16bf b1 = ld16(wrow1 + bo, wrow1 + bo + 16);
      acc00 = __builtin_amdgcn_wmma_f32_16x16x32_bf16(false, a0, false, b0, (short)0, acc00, false, false);
      acc01 = __builtin_amdgcn_wmma_f32_16x16x32_bf16(false, a0, false, b1, (short)0, acc01, false, false);
      acc10 = __builtin_amdgcn_wmma_f32_16x16x32_bf16(false, a1, false, b0, (short)0, acc10, false, false);
      acc11 = __builtin_amdgcn_wmma_f32_16x16x32_bf16(false, a1, false, b1, (short)0, acc11, false, false);
    }
  }
  // Segment 2 (optional): K2 over A2 and W[:, K1:K1+K2]
  if (K2 > 0) {
    const __bf16* arow0 = A2 + (size_t)(rowBase + r) * K2;
    const __bf16* arow1 = arow0 + (size_t)16 * K2;
    const __bf16* wrow0 = W + (size_t)(colBase + r) * ldW + K1;
    const __bf16* wrow1 = wrow0 + (size_t)16 * ldW;
    for (int kb = 0; kb < K2; kb += 32) {
      int ao = kb + half * 8;
      int bo = kb + half * 8;
      v16bf a0 = ld16(arow0 + ao, arow0 + ao + 16);
      v16bf a1 = ld16(arow1 + ao, arow1 + ao + 16);
      v16bf b0 = ld16(wrow0 + bo, wrow0 + bo + 16);
      v16bf b1 = ld16(wrow1 + bo, wrow1 + bo + 16);
      acc00 = __builtin_amdgcn_wmma_f32_16x16x32_bf16(false, a0, false, b0, (short)0, acc00, false, false);
      acc01 = __builtin_amdgcn_wmma_f32_16x16x32_bf16(false, a0, false, b1, (short)0, acc01, false, false);
      acc10 = __builtin_amdgcn_wmma_f32_16x16x32_bf16(false, a1, false, b0, (short)0, acc10, false, false);
      acc11 = __builtin_amdgcn_wmma_f32_16x16x32_bf16(false, a1, false, b1, (short)0, acc11, false, false);
    }
  }

  // C/D layout: lane<16 -> N=lane, M=v ; lane>=16 -> N=lane-16, M=8+v
  int col0 = colBase + r;
  int col1 = col0 + 16;
  float bv0 = bias ? bias[col0] : 0.0f;
  float bv1 = bias ? bias[col1] : 0.0f;
#pragma unroll
  for (int v = 0; v < 8; ++v) {
    int row0 = rowBase + half * 8 + v;
    int row1 = row0 + 16;
    float r00 = scale * (acc00[v] + bv0);
    float r01 = scale * (acc01[v] + bv1);
    float r10 = scale * (acc10[v] + bv0);
    float r11 = scale * (acc11[v] + bv1);
    if (addsrc) {
      r00 += addsrc[(size_t)row0 * addld + col0];
      r01 += addsrc[(size_t)row0 * addld + col1];
      r10 += addsrc[(size_t)row1 * addld + col0];
      r11 += addsrc[(size_t)row1 * addld + col1];
    }
    out[(size_t)row0 * outld + col0] = r00;
    out[(size_t)row0 * outld + col1] = r01;
    out[(size_t)row1 * outld + col0] = r10;
    out[(size_t)row1 * outld + col1] = r11;
  }
}

// ---------------------------------------------------------------------------
// LayerNorm over n (=2048) per row -> bf16 output. block = 256 threads, 1 row.
// ---------------------------------------------------------------------------
__global__ void k_layernorm(const float* __restrict__ in, const float* __restrict__ w,
                            const float* __restrict__ bsh, __bf16* __restrict__ outbf, int n) {
  __shared__ float sm1[8];
  __shared__ float sm2[8];
  int row = blockIdx.x;
  int tid = threadIdx.x;
  const float* x = in + (size_t)row * n;
  float vals[8];
  int per = n >> 8;  // n / 256  (<= 8)
  float s = 0.f;
  for (int i = 0; i < per; ++i) { float v = x[tid + (i << 8)]; vals[i] = v; s += v; }
  for (int o = 16; o > 0; o >>= 1) s += __shfl_xor(s, o, 32);
  if ((tid & 31) == 0) sm1[tid >> 5] = s;
  __syncthreads();
  float tot = 0.f;
  for (int i = 0; i < 8; ++i) tot += sm1[i];
  float mu = tot / (float)n;
  float s2 = 0.f;
  for (int i = 0; i < per; ++i) { float d = vals[i] - mu; s2 += d * d; }
  for (int o = 16; o > 0; o >>= 1) s2 += __shfl_xor(s2, o, 32);
  if ((tid & 31) == 0) sm2[tid >> 5] = s2;
  __syncthreads();
  float tot2 = 0.f;
  for (int i = 0; i < 8; ++i) tot2 += sm2[i];
  float rstd = rsqrtf(tot2 / (float)n + 1e-5f);
  for (int i = 0; i < per; ++i) {
    int j = tid + (i << 8);
    outbf[(size_t)row * n + j] = (__bf16)((vals[i] - mu) * rstd * w[j] + bsh[j]);
  }
}

// ---------------------------------------------------------------------------
// LTC cell update + layernorm over M=512, fused. block = 256 threads, 1 row.
// hf : f32 state (read old, write new); hb : bf16 copy of new state.
// outdst (optional): outs[:, t, :] slice, stride outld per batch row.
// ---------------------------------------------------------------------------
__global__ void k_ltc_update(const float* __restrict__ zu, const float* __restrict__ zr,
                             const float* __restrict__ zc, const float* __restrict__ tau,
                             const float* __restrict__ mnw, const float* __restrict__ mnb,
                             float* __restrict__ hf, __bf16* __restrict__ hb,
                             float* __restrict__ outdst, long outld) {
  __shared__ float sm1[8];
  __shared__ float sm2[8];
  int b = blockIdx.x;
  int tid = threadIdx.x;
  float hn[2];
  float s = 0.f;
#pragma unroll
  for (int i = 0; i < 2; ++i) {
    int j = tid + (i << 8);
    size_t idx = (size_t)b * MM + j;
    float u = 1.f / (1.f + expf(-zu[idx]));
    float r = 1.f / (1.f + expf(-zr[idx]));
    float cand = tanhf(zc[idx]);
    float tv = tau[j];
    float sp = (tv > 20.f) ? tv : log1pf(expf(tv));
    sp = fminf(fmaxf(sp, 0.1f), 10.f);
    float decay = expf(-1.f / sp) * u;
    float h = hf[idx];
    hn[i] = decay * (h * r) + (1.f - decay) * cand;
    s += hn[i];
  }
  for (int o = 16; o > 0; o >>= 1) s += __shfl_xor(s, o, 32);
  if ((tid & 31) == 0) sm1[tid >> 5] = s;
  __syncthreads();
  float tot = 0.f;
  for (int i = 0; i < 8; ++i) tot += sm1[i];
  float mu = tot / (float)MM;
  float s2 = 0.f;
#pragma unroll
  for (int i = 0; i < 2; ++i) { float d = hn[i] - mu; s2 += d * d; }
  for (int o = 16; o > 0; o >>= 1) s2 += __shfl_xor(s2, o, 32);
  if ((tid & 31) == 0) sm2[tid >> 5] = s2;
  __syncthreads();
  float tot2 = 0.f;
  for (int i = 0; i < 8; ++i) tot2 += sm2[i];
  float rstd = rsqrtf(tot2 / (float)MM + 1e-5f);
#pragma unroll
  for (int i = 0; i < 2; ++i) {
    int j = tid + (i << 8);
    size_t idx = (size_t)b * MM + j;
    float v = (hn[i] - mu) * rstd * mnw[j] + mnb[j];
    hf[idx] = v;
    hb[idx] = (__bf16)v;
    if (outdst) outdst[(size_t)b * outld + j] = v;
  }
}

// ---------------------------------------------------------------------------
extern "C" void kernel_launch(void* const* d_in, const int* in_sizes, int n_in,
                              void* d_out, int out_size, void* d_ws, size_t ws_size,
                              hipStream_t stream) {
  (void)in_sizes; (void)n_in; (void)out_size; (void)ws_size;
  const float* x      = (const float*)d_in[0];
  const float* hidden = (const float*)d_in[1];
  const float* W_in   = (const float*)d_in[2];
  const float* b_in   = (const float*)d_in[3];
  const float* cell_W = (const float*)d_in[4];
  const float* cell_b = (const float*)d_in[5];
  const float* upd_W  = (const float*)d_in[6];
  const float* upd_b  = (const float*)d_in[7];
  const float* rst_W  = (const float*)d_in[8];
  const float* rst_b  = (const float*)d_in[9];
  const float* tau    = (const float*)d_in[10];
  const float* lc_W   = (const float*)d_in[11];
  const float* lc_b   = (const float*)d_in[12];
  const float* sk_W   = (const float*)d_in[13];
  const float* sk_b   = (const float*)d_in[14];
  const float* fb_W   = (const float*)d_in[15];
  const float* fb_b   = (const float*)d_in[16];
  const float* ln_w   = (const float*)d_in[17];
  const float* ln_b   = (const float*)d_in[18];
  const float* mn_w   = (const float*)d_in[19];
  const float* mn_b   = (const float*)d_in[20];
  float* out = (float*)d_out;

  // ---- workspace carve-up (256B aligned) ----
  char* base = (char*)d_ws;
  size_t off = 0;
  auto carve = [&](size_t bytes) -> char* {
    char* p = base + off;
    off = (off + bytes + 255) & ~(size_t)255;
    return p;
  };
  __bf16* xbf    = (__bf16*)carve((size_t)BB * TT * NIN * 2);
  __bf16* winbf  = (__bf16*)carve((size_t)HH * NIN * 2);
  float*  xp     = (float*) carve((size_t)BB * TT * HH * 4);
  __bf16* updbf  = (__bf16*)carve((size_t)LLAY * MM * HM * 2);
  __bf16* rstbf  = (__bf16*)carve((size_t)LLAY * MM * HM * 2);
  __bf16* cellbf = (__bf16*)carve((size_t)LLAY * MM * HH * 2);
  __bf16* fbbf   = (__bf16*)carve((size_t)HH * MM * 2);
  __bf16* cw1bf  = (__bf16*)carve((size_t)HH * MM * 2);
  float*  cb1    = (float*) carve((size_t)HH * 4);
  float*  li0    = (float*) carve((size_t)BB * HH * 4);
  __bf16* li0b   = (__bf16*)carve((size_t)BB * HH * 2);
  float*  li1    = (float*) carve((size_t)BB * HH * 4);
  __bf16* li1b   = (__bf16*)carve((size_t)BB * HH * 2);
  float*  zu     = (float*) carve((size_t)BB * MM * 4);
  float*  zr     = (float*) carve((size_t)BB * MM * 4);
  float*  zc     = (float*) carve((size_t)BB * MM * 4);
  float*  h0f    = (float*) carve((size_t)BB * MM * 4);
  float*  h1f    = (float*) carve((size_t)BB * MM * 4);
  __bf16* h0b    = (__bf16*)carve((size_t)BB * MM * 2);
  __bf16* h1b    = (__bf16*)carve((size_t)BB * MM * 2);

  auto cvt = [&](const float* src, __bf16* dst, int n) {
    k_f32_to_bf16<<<(n + 255) / 256, 256, 0, stream>>>(src, dst, n);
  };

  // ---- one-time (per launch) conversions / weight combining ----
  cvt(x, xbf, BB * TT * NIN);
  cvt(W_in, winbf, HH * NIN);
  cvt(upd_W, updbf, LLAY * MM * HM);
  cvt(rst_W, rstbf, LLAY * MM * HM);
  cvt(cell_W, cellbf, LLAY * MM * HH);
  cvt(fb_W, fbbf, HH * MM);
  k_add_to_bf16<<<(HH * MM + 255) / 256, 256, 0, stream>>>(sk_W, lc_W, cw1bf, HH * MM);
  k_add_f32<<<(HH + 255) / 256, 256, 0, stream>>>(sk_b, lc_b, cb1, HH);
  cvt(hidden, h0b, BB * MM);
  cvt(hidden + BB * MM, h1b, BB * MM);
  hipMemcpyAsync(h0f, hidden, (size_t)BB * MM * 4, hipMemcpyDeviceToDevice, stream);
  hipMemcpyAsync(h1f, hidden + BB * MM, (size_t)BB * MM * 4, hipMemcpyDeviceToDevice, stream);

  // ---- input projection: xp[BT,H] = x[BT,IN] @ W_in[H,IN]^T + b_in ----
  {
    int rt2 = (BB * TT) / 32, ct2 = HH / 32;  // 512 x 64 super-tiles
    int blocks = (rt2 * ct2 + 7) / 8;
    k_gemm<<<blocks, 256, 0, stream>>>(xbf, NIN, nullptr, 0, winbf, NIN,
                                       b_in, 1.0f, nullptr, 0, xp, HH, rt2, ct2);
  }

  const int rtB = BB / 32;                      // 2 row super-tiles
  const int ctH = HH / 32, ctM = MM / 32;       // 64 / 16 col super-tiles
  const int blkH = (rtB * ctH + 7) / 8;         // 16 blocks
  const int blkM = (rtB * ctM + 7) / 8;         // 4 blocks

  // ---- sequential scan over time ----
  for (int t = 0; t < TT; ++t) {
    // layer 0 input: li0 = xp[:,t,:] + (t>0 ? 0.1*(h1_prev @ fb_W^T + fb_b) : 0)
    k_gemm<<<blkH, 256, 0, stream>>>(h1b, MM, nullptr, 0, fbbf, MM,
                                     fb_b, (t > 0) ? 0.1f : 0.0f,
                                     xp + (size_t)t * HH, TT * HH,
                                     li0, HH, rtB, ctH);
    k_layernorm<<<BB, 256, 0, stream>>>(li0, ln_w, ln_b, li0b, HH);
    // gates (two-segment K: li over H, h over M; concat never materialized)
    k_gemm<<<blkM, 256, 0, stream>>>(li0b, HH, h0b, MM, updbf, HM,
                                     upd_b, 1.0f, nullptr, 0, zu, MM, rtB, ctM);
    k_gemm<<<blkM, 256, 0, stream>>>(li0b, HH, h0b, MM, rstbf, HM,
                                     rst_b, 1.0f, nullptr, 0, zr, MM, rtB, ctM);
    k_gemm<<<blkM, 256, 0, stream>>>(li0b, HH, nullptr, 0, cellbf, HH,
                                     cell_b, 1.0f, nullptr, 0, zc, MM, rtB, ctM);
    k_ltc_update<<<BB, 256, 0, stream>>>(zu, zr, zc, tau, mn_w, mn_b,
                                         h0f, h0b, nullptr, 0);

    // layer 1 input: li1 = hn0 @ (sk_W + lc_W)^T + (sk_b + lc_b)
    k_gemm<<<blkH, 256, 0, stream>>>(h0b, MM, nullptr, 0, cw1bf, MM,
                                     cb1, 1.0f, nullptr, 0, li1, HH, rtB, ctH);
    k_layernorm<<<BB, 256, 0, stream>>>(li1, ln_w + HH, ln_b + HH, li1b, HH);
    k_gemm<<<blkM, 256, 0, stream>>>(li1b, HH, h1b, MM, updbf + (size_t)MM * HM, HM,
                                     upd_b + MM, 1.0f, nullptr, 0, zu, MM, rtB, ctM);
    k_gemm<<<blkM, 256, 0, stream>>>(li1b, HH, h1b, MM, rstbf + (size_t)MM * HM, HM,
                                     rst_b + MM, 1.0f, nullptr, 0, zr, MM, rtB, ctM);
    k_gemm<<<blkM, 256, 0, stream>>>(li1b, HH, nullptr, 0, cellbf + (size_t)MM * HH, HH,
                                     cell_b + MM, 1.0f, nullptr, 0, zc, MM, rtB, ctM);
    // update h1 and emit outs[:, t, :]
    k_ltc_update<<<BB, 256, 0, stream>>>(zu, zr, zc, tau + MM, mn_w + MM, mn_b + MM,
                                         h1f, h1b, out + (size_t)t * MM, (long)TT * MM);
  }

  // ---- final hidden states hT [L,B,M] appended after outs [B,T,M] ----
  float* hT = out + (size_t)BB * TT * MM;
  hipMemcpyAsync(hT, h0f, (size_t)BB * MM * 4, hipMemcpyDeviceToDevice, stream);
  hipMemcpyAsync(hT + (size_t)BB * MM, h1f, (size_t)BB * MM * 4, hipMemcpyDeviceToDevice, stream);
}
